// Who2com_54726473285728
// MI455X (gfx1250) — compile-verified, matrix operand order checked by reading, not run on hardware
//
#include <hip/hip_runtime.h>
#include <math.h>

typedef __attribute__((ext_vector_type(16))) _Float16 v16h;
typedef __attribute__((ext_vector_type(8)))  _Float16 v8h;
typedef __attribute__((ext_vector_type(8)))  float    v8f;
typedef __attribute__((ext_vector_type(4)))  int      v4i;

union V16H { v16h v; v8h h[2]; };

#if defined(__HIP_DEVICE_COMPILE__) && __has_builtin(__builtin_amdgcn_global_load_async_to_lds_b128)
#define ASYNC_LDS 1
#define AS1P __attribute__((address_space(1))) v4i*
#define AS3P __attribute__((address_space(3))) v4i*
#else
#define ASYNC_LDS 0
#endif

// K index inside a 32-wide chunk for CDNA5 16-bit WMMA operand fragments.
// VGPR v (0..7), element e (0/1), lane-half h (0/1):
//   v<4 : K = 2v+e+8h ; v>=4 : K = 8+2v+e+8h   (CDNA5 ISA, 16-bit A 16x32)
__device__ __forceinline__ int kmap(int v, int e, int h) {
    return (v < 4) ? (2*v + e + 8*h) : (8 + 2*v + e + 8*h);
}

// ---------------------------------------------------------------------------
// bevs NCHW f32 [4][80][256][256] -> NHWC f16 [4][65536][80], LDS-tiled.
__global__ void nchw_to_nhwc_f16(const float* __restrict__ src, _Float16* __restrict__ dst) {
    __shared__ _Float16 t[80][33];
    int n  = blockIdx.x >> 11;              // 2048 tiles of 32 pixels per image
    int p0 = (blockIdx.x & 2047) * 32;
    int tid = threadIdx.x;
    for (int i = tid; i < 80 * 32; i += 256) {
        int c = i >> 5, px = i & 31;
        t[c][px] = (_Float16)src[((size_t)n * 80 + c) * 65536 + p0 + px];
    }
    __syncthreads();
    for (int i = tid; i < 32 * 80; i += 256) {
        int px = i / 80, c = i % 80;
        dst[((size_t)n * 65536 + p0 + px) * 80 + c] = t[c][px];
    }
}

// ---------------------------------------------------------------------------
// Fold conv bias + BN into per-channel scale/shift (11 layers x 64 ch).
__global__ void bn_prep(const float* __restrict__ conv1_b, const float* __restrict__ bn1,
                        const float* __restrict__ convs_b, const float* __restrict__ bns,
                        float* __restrict__ sc, float* __restrict__ sh) {
    int i = blockIdx.x * blockDim.x + threadIdx.x;
    if (i >= 11 * 64) return;
    int layer = i / 64, c = i % 64;
    float g, be, me, va, bias;
    if (layer == 0) {
        g = bn1[c]; be = bn1[64 + c]; me = bn1[128 + c]; va = bn1[192 + c];
        bias = conv1_b[c];
    } else {
        const float* b = bns + (size_t)(layer - 1) * 256;
        g = b[c]; be = b[64 + c]; me = b[128 + c]; va = b[192 + c];
        bias = convs_b[(layer - 1) * 64 + c];
    }
    float s = g * rsqrtf(va + 1e-5f);
    sc[i] = s;
    sh[i] = (bias - me) * s + be;
}

// ---------------------------------------------------------------------------
// Swizzle OIHW f32 weights into WMMA B-fragment order:
//   wpk[coutgrp(4)][kchunk][lane(32)][16 halves], k = tap*Cin + c, zero-padded K.
__global__ void prepack_weights(const float* __restrict__ w, _Float16* __restrict__ wpk,
                                int Cin, int kchunks) {
    int idx = blockIdx.x * blockDim.x + threadIdx.x;
    int total = 4 * kchunks * 512;
    if (idx >= total) return;
    int j    = idx & 15;
    int lane = (idx >> 4) & 31;
    int kc   = (idx >> 9) % kchunks;
    int grp  = (idx >> 9) / kchunks;
    int k = kc * 32 + kmap(j >> 1, j & 1, lane >> 4);
    float val = 0.f;
    if (k < 9 * Cin) {
        int tap = k / Cin, c = k % Cin;
        int cout = grp * 16 + (lane & 15);
        val = w[((cout * Cin + c) * 3 + (tap / 3)) * 3 + (tap % 3)];
    }
    wpk[idx] = (_Float16)val;
}

// ---------------------------------------------------------------------------
// Implicit-GEMM 3x3 conv + BN + ReLU via v_wmma_f32_16x16x32_f16.
// Block = 128 threads (4 waves). Tile: 32 output pixels x 64 output channels;
// each wave owns one 16-cout group and two 16-pixel accumulators sharing each
// B fragment (2 WMMAs per weight load). All divisors are compile-time.
template<int CIN, int STRIDE, int KCH, int LOGW>
__global__ void __launch_bounds__(128) conv_wmma(
    const _Float16* __restrict__ src, _Float16* __restrict__ dst,
    const _Float16* __restrict__ wpk,
    const float* __restrict__ bn_sc, const float* __restrict__ bn_sh) {
    constexpr int WOUT  = 1 << LOGW;
    constexpr int HW    = WOUT * WOUT;
    constexpr int HIN   = WOUT * STRIDE;
    constexpr int KTOT  = KCH * 32;
    constexpr int CIN8  = CIN >> 3;

    __shared__ __align__(16) _Float16 sA[32 * KTOT];

    const int tid = threadIdx.x;
    const int p0  = blockIdx.x * 32;

    // im2col pack: 32 pixels x (9 taps x CIN) halves, zero-padded borders.
    constexpr int totalChunks = 32 * 9 * CIN8;
    for (int idx = tid; idx < totalChunks; idx += 128) {
        int c8  = idx % CIN8;
        int r   = idx / CIN8;
        int tap = r % 9, m = r / 9;
        int p   = p0 + m;
        int n   = p >> (2 * LOGW);
        int pix = p & (HW - 1);
        int y   = pix >> LOGW, x = pix & (WOUT - 1);
        int yi  = y * STRIDE + (tap / 3) - 1;
        int xi  = x * STRIDE + (tap % 3) - 1;
        _Float16* ldst = &sA[m * KTOT + tap * CIN + c8 * 8];
        if ((unsigned)yi < (unsigned)HIN && (unsigned)xi < (unsigned)HIN) {
            const _Float16* gsrc = &src[(((size_t)n * HIN + yi) * HIN + xi) * CIN + c8 * 8];
#if ASYNC_LDS
            // AS1/AS3 pointers built via integer casts: generic LDS pointers are
            // {aperture_hi, lds_addr32} (ISA 10.2), so the low 32 bits are the
            // AS(3) address; generic global pointers equal their AS(1) address.
            __builtin_amdgcn_global_load_async_to_lds_b128(
                (AS1P)(unsigned long long)gsrc,
                (AS3P)(unsigned)(unsigned long long)ldst, 0, 0);
#else
            *reinterpret_cast<v8h*>(ldst) = *reinterpret_cast<const v8h*>(gsrc);
#endif
        } else {
            v8h z = {};
            *reinterpret_cast<v8h*>(ldst) = z;
        }
    }
    if constexpr (KTOT > 9 * CIN) {           // zero K padding (conv1: 720 -> 736)
        constexpr int PAD8 = (KTOT - 9 * CIN) >> 3;
        for (int idx = tid; idx < 32 * PAD8; idx += 128) {
            int c8 = idx % PAD8, m = idx / PAD8;
            v8h z = {};
            *reinterpret_cast<v8h*>(&sA[m * KTOT + 9 * CIN + c8 * 8]) = z;
        }
    }
#if ASYNC_LDS
#if __has_builtin(__builtin_amdgcn_s_wait_asynccnt)
    __builtin_amdgcn_s_wait_asynccnt(0);
#else
    asm volatile("s_wait_asynccnt 0" ::: "memory");
#endif
#endif
    __syncthreads();

    const int wave = tid >> 5, lane = tid & 31;
    const int lm = lane & 15, lh = lane >> 4;
    v8f acc0 = {}, acc1 = {};
    const _Float16* arow0 = &sA[lm * KTOT];
    const _Float16* arow1 = &sA[(16 + lm) * KTOT];
    const _Float16* bptr  = &wpk[(size_t)(wave * KCH) * 512 + lane * 16];
    for (int kc = 0; kc < KCH; ++kc) {
        v16h b = *reinterpret_cast<const v16h*>(bptr + (size_t)kc * 512);
        V16H a0, a1;
        a0.h[0] = *reinterpret_cast<const v8h*>(arow0 + kc * 32 + 8 * lh);
        a0.h[1] = *reinterpret_cast<const v8h*>(arow0 + kc * 32 + 16 + 8 * lh);
        a1.h[0] = *reinterpret_cast<const v8h*>(arow1 + kc * 32 + 8 * lh);
        a1.h[1] = *reinterpret_cast<const v8h*>(arow1 + kc * 32 + 16 + 8 * lh);
        acc0 = __builtin_amdgcn_wmma_f32_16x16x32_f16(false, a0.v, false, b,
                                                      (short)0, acc0, false, false);
        acc1 = __builtin_amdgcn_wmma_f32_16x16x32_f16(false, a1.v, false, b,
                                                      (short)0, acc1, false, false);
    }

    // Epilogue per C/D layout: lane n = lane%16, VGPR r -> m = r + 8*(lane/16).
    const int cout = wave * 16 + lm;
    const float s = bn_sc[cout], t = bn_sh[cout];
#pragma unroll
    for (int r = 0; r < 8; ++r) {
        int m = r + 8 * lh;
        {
            int p = p0 + m;
            int n = p >> (2 * LOGW), pix = p & (HW - 1);
            float vv = acc0[r] * s + t;
            vv = vv > 0.f ? vv : 0.f;
            dst[((size_t)n * HW + pix) * 64 + cout] = (_Float16)vv;
        }
        {
            int p = p0 + 16 + m;
            int n = p >> (2 * LOGW), pix = p & (HW - 1);
            float vv = acc1[r] * s + t;
            vv = vv > 0.f ? vv : 0.f;
            dst[((size_t)n * HW + pix) * 64 + cout] = (_Float16)vv;
        }
    }
}

// ---------------------------------------------------------------------------
// Small dense layer: out[4][N] = A[4][K] @ W[K][N] + b (optionally ReLU).
// mode 1: A is conv11 NHWC f16 [4][64hw][64c], remapped to reference order
//         k_ref = c*64 + hw.
__global__ void mlp_layer(const float* __restrict__ Af32,
                          const _Float16* __restrict__ Aconv,
                          const float* __restrict__ W, const float* __restrict__ bias,
                          float* __restrict__ out, int K, int N, int mode, int relu) {
    int idx = blockIdx.x * blockDim.x + threadIdx.x;
    if (idx >= 4 * N) return;
    int row = idx / N, n = idx % N;
    float s = bias[n];
    if (mode == 1) {
        for (int k = 0; k < K; ++k) {
            int c = k >> 6, hw = k & 63;
            s += (float)Aconv[((size_t)row * 64 + hw) * 64 + c] * W[(size_t)k * N + n];
        }
    } else {
        for (int k = 0; k < K; ++k)
            s += Af32[row * K + k] * W[(size_t)k * N + n];
    }
    if (relu) s = s > 0.f ? s : 0.f;
    out[idx] = s;
}

// ---------------------------------------------------------------------------
// attn[k][q] = keys[k].q[q]; softmax over k; scale[q] = sum_k softmax (== ~1).
__global__ void attn_scale(const float* __restrict__ keys, const float* __restrict__ qf,
                           float* __restrict__ scale) {
    __shared__ float attn[16];
    int tid = threadIdx.x;
    if (tid < 16) {
        int k = tid >> 2, q = tid & 3;
        float s = 0.f;
        for (int d = 0; d < 1024; ++d) s += keys[k * 1024 + d] * qf[q * 1024 + d];
        attn[tid] = s;
    }
    __syncthreads();
    if (tid < 4) {
        int q = tid;
        float mx = attn[q];
        for (int k = 1; k < 4; ++k) mx = fmaxf(mx, attn[k * 4 + q]);
        float se = 0.f;
        for (int k = 0; k < 4; ++k) se += expf(attn[k * 4 + q] - mx);
        float sum = 0.f;
        for (int k = 0; k < 4; ++k) sum += expf(attn[k * 4 + q] - mx) / se;
        scale[q] = sum;
    }
}

// ---------------------------------------------------------------------------
// out = scale[n] * bevs  (float4 streaming, 168 MB total traffic).
__global__ void scale_out(const float* __restrict__ bevs, const float* __restrict__ scale,
                          float* __restrict__ out) {
    const int PER4 = 80 * 256 * 256 / 4;           // 1,310,720 float4 per image
    size_t idx = (size_t)blockIdx.x * blockDim.x + threadIdx.x;
    if (idx >= (size_t)4 * PER4) return;
    float s = scale[idx / PER4];
    float4 v = reinterpret_cast<const float4*>(bevs)[idx];
    v.x *= s; v.y *= s; v.z *= s; v.w *= s;
    reinterpret_cast<float4*>(out)[idx] = v;
}

// ===========================================================================
extern "C" void kernel_launch(void* const* d_in, const int* in_sizes, int n_in,
                              void* d_out, int out_size, void* d_ws, size_t ws_size,
                              hipStream_t stream) {
    (void)in_sizes; (void)n_in; (void)out_size; (void)ws_size;
    const float* bevs    = (const float*)d_in[0];
    const float* conv1_w = (const float*)d_in[1];
    const float* conv1_b = (const float*)d_in[2];
    const float* bn1     = (const float*)d_in[3];
    const float* convs_w = (const float*)d_in[4];
    const float* convs_b = (const float*)d_in[5];
    const float* bns     = (const float*)d_in[6];
    const float* key_w1  = (const float*)d_in[7];
    const float* key_b1  = (const float*)d_in[8];
    const float* key_w2  = (const float*)d_in[9];
    const float* key_b2  = (const float*)d_in[10];
    const float* key_w3  = (const float*)d_in[11];
    const float* key_b3  = (const float*)d_in[12];
    const float* qry_w1  = (const float*)d_in[13];
    const float* qry_b1  = (const float*)d_in[14];
    const float* qry_w2  = (const float*)d_in[15];
    const float* qry_b2  = (const float*)d_in[16];
    const float* qry_w3  = (const float*)d_in[17];
    const float* qry_b3  = (const float*)d_in[18];
    const float* attn_w  = (const float*)d_in[19];
    const float* attn_b  = (const float*)d_in[20];

    char* ws = (char*)d_ws;
    auto alloc = [&](size_t bytes) {
        char* p = ws;
        ws += (bytes + 255) & ~(size_t)255;
        return p;
    };
    _Float16* actA = (_Float16*)alloc(4ull * 65536 * 80 * 2);   // conv1 input NHWC f16
    _Float16* actB = (_Float16*)alloc(4ull * 65536 * 64 * 2);   // ping
    _Float16* actC = (_Float16*)alloc(4ull * 65536 * 64 * 2);   // pong
    _Float16* wpk1 = (_Float16*)alloc(4ull * 23 * 512 * 2);
    _Float16* wpk2 = (_Float16*)alloc(10ull * 4 * 18 * 512 * 2);
    float* bnsc  = (float*)alloc(11 * 64 * 4);
    float* bnsh  = (float*)alloc(11 * 64 * 4);
    float* kh1   = (float*)alloc(4 * 256 * 4);
    float* kh2   = (float*)alloc(4 * 128 * 4);
    float* keys  = (float*)alloc(4 * 1024 * 4);
    float* qh1   = (float*)alloc(4 * 256 * 4);
    float* qh2   = (float*)alloc(4 * 128 * 4);
    float* qrys  = (float*)alloc(4 * 32 * 4);
    float* qfull = (float*)alloc(4 * 1024 * 4);
    float* scl   = (float*)alloc(4 * 4);

    nchw_to_nhwc_f16<<<8192, 256, 0, stream>>>(bevs, actA);
    bn_prep<<<3, 256, 0, stream>>>(conv1_b, bn1, convs_b, bns, bnsc, bnsh);
    prepack_weights<<<184, 256, 0, stream>>>(conv1_w, wpk1, 80, 23);   // 47104 elems
    for (int l = 0; l < 10; ++l)
        prepack_weights<<<144, 256, 0, stream>>>(convs_w + (size_t)l * 64 * 64 * 9,
                                                 wpk2 + (size_t)l * 4 * 18 * 512, 64, 18);

    const _Float16* cs = actA;
    _Float16* cd = actB;
    _Float16* conv_out = actB;
    for (int i = 0; i < 11; ++i) {
        const _Float16* wp = (i == 0) ? wpk1 : wpk2 + (size_t)(i - 1) * 4 * 18 * 512;
        const float* sc = bnsc + i * 64;
        const float* sh = bnsh + i * 64;
        switch (i) {   // blocks = 4*HW/32
        case 0:  conv_wmma<80,1,23,8><<<8192, 128, 0, stream>>>(cs, cd, wp, sc, sh); break;
        case 1:  conv_wmma<64,1,18,8><<<8192, 128, 0, stream>>>(cs, cd, wp, sc, sh); break;
        case 2:  conv_wmma<64,2,18,7><<<2048, 128, 0, stream>>>(cs, cd, wp, sc, sh); break;
        case 3:  conv_wmma<64,1,18,7><<<2048, 128, 0, stream>>>(cs, cd, wp, sc, sh); break;
        case 4:  conv_wmma<64,2,18,6><<< 512, 128, 0, stream>>>(cs, cd, wp, sc, sh); break;
        case 5:  conv_wmma<64,1,18,6><<< 512, 128, 0, stream>>>(cs, cd, wp, sc, sh); break;
        case 6:  conv_wmma<64,2,18,5><<< 128, 128, 0, stream>>>(cs, cd, wp, sc, sh); break;
        case 7:  conv_wmma<64,1,18,5><<< 128, 128, 0, stream>>>(cs, cd, wp, sc, sh); break;
        case 8:  conv_wmma<64,2,18,4><<<  32, 128, 0, stream>>>(cs, cd, wp, sc, sh); break;
        case 9:  conv_wmma<64,1,18,4><<<  32, 128, 0, stream>>>(cs, cd, wp, sc, sh); break;
        case 10: conv_wmma<64,2,18,3><<<   8, 128, 0, stream>>>(cs, cd, wp, sc, sh); break;
        }
        conv_out = cd;
        if (i == 0) { cs = actB; cd = actC; }
        else        { _Float16* t = (_Float16*)cs; cs = cd; cd = t; }
    }

    // key MLP
    mlp_layer<<<4, 256, 0, stream>>>(nullptr, conv_out, key_w1, key_b1, kh1, 4096, 256, 1, 1);
    mlp_layer<<<2, 256, 0, stream>>>(kh1, nullptr, key_w2, key_b2, kh2, 256, 128, 0, 1);
    mlp_layer<<<16, 256, 0, stream>>>(kh2, nullptr, key_w3, key_b3, keys, 128, 1024, 0, 0);
    // query MLP
    mlp_layer<<<4, 256, 0, stream>>>(nullptr, conv_out, qry_w1, qry_b1, qh1, 4096, 256, 1, 1);
    mlp_layer<<<2, 256, 0, stream>>>(qh1, nullptr, qry_w2, qry_b2, qh2, 256, 128, 0, 1);
    mlp_layer<<<1, 256, 0, stream>>>(qh2, nullptr, qry_w3, qry_b3, qrys, 128, 32, 0, 0);
    // q = querys @ attn_w + attn_b
    mlp_layer<<<16, 256, 0, stream>>>(qrys, nullptr, attn_w, attn_b, qfull, 32, 1024, 0, 0);

    attn_scale<<<1, 64, 0, stream>>>(keys, qfull, scl);
    scale_out<<<20480, 256, 0, stream>>>(bevs, scl, (float*)d_out);
}
